// CrossAttnBlock_59158879535168
// MI455X (gfx1250) — matrix operand order repaired and checked by reading, not compiled, not run on hardware
//
#include <hip/hip_runtime.h>
#include <hip/hip_bf16.h>
#include <stdint.h>

// ---------------------------------------------------------------------------
// CrossAttnBlock for MI455X (gfx1250): GN -> QKV GEMM -> flash attention -> proj
// Matrix math via v_wmma_f32_16x16x32_f16; attention K/V staged into LDS with
// gfx1250 async global->LDS DMA (ASYNCcnt) and double buffering.
// ---------------------------------------------------------------------------

typedef __attribute__((ext_vector_type(16))) _Float16 v16h;
typedef __attribute__((ext_vector_type(8)))  _Float16 v8h;
typedef __attribute__((ext_vector_type(8)))  float    v8f;

#define BATCH 8
#define CH    256
#define HW    4096          // 64*64
#define NG    32            // groups
#define CG    8             // channels per group (256/32)
#define NROWS (BATCH * HW)  // 32768 pixel rows
#define ATT_SCALE 0.0625f   // 256^-0.5
#define GN_EPS 1e-5f

#define KSTRIDE 264         // halves per K-row in LDS (padded; 528B, 16B aligned)
#define VSTRIDE 40          // halves per V-row in LDS (padded; 80B, 16B aligned)

__device__ __forceinline__ v8f wmma_f16(v16h a, v16h b, v8f c) {
    return __builtin_amdgcn_wmma_f32_16x16x32_f16(false, a, false, b,
                                                  (short)0, c, false, false);
}

__device__ __forceinline__ v8f vzero8f() {
    v8f z;
#pragma unroll
    for (int i = 0; i < 8; ++i) z[i] = 0.0f;
    return z;
}

// Build a contiguous 16-half B/A fragment from two 16B-aligned 8-half loads.
__device__ __forceinline__ v16h ld16h(const _Float16* p) {
    v8h lo = *(const v8h*)p;
    v8h hi = *(const v8h*)(p + 8);
    v16h r;
#pragma unroll
    for (int i = 0; i < 8; ++i) { r[i] = lo[i]; r[8 + i] = hi[i]; }
    return r;
}

// gfx1250 async DMA: 16 bytes global -> LDS, tracked by ASYNCcnt (no VGPR data).
// LDS destination is the low 32 bits of the generic pointer (LDS-relative).
__device__ __forceinline__ void async_g2l_b128(void* lds_dst, const void* gsrc) {
    uint32_t loff = (uint32_t)(uintptr_t)lds_dst;
    asm volatile("global_load_async_to_lds_b128 %0, %1, off"
                 :: "v"(loff), "v"(gsrc) : "memory");
}

__device__ __forceinline__ void wait_async0() {
    asm volatile("s_wait_asynccnt 0x0" ::: "memory");
}

// ---------------------------------------------------------------------------
// Weight fp32 -> fp16 conversion
// ---------------------------------------------------------------------------
__global__ void cvt_w_kernel(const float* __restrict__ w, _Float16* __restrict__ o, int n) {
    int i = blockIdx.x * blockDim.x + threadIdx.x;
    if (i < n) o[i] = (_Float16)w[i];
}

// ---------------------------------------------------------------------------
// GroupNorm; writes f16 output TRANSPOSED to [B][pixel][C].
// grid.x = B*NG, grid.y = 2 (x / target)
// ---------------------------------------------------------------------------
__global__ void __launch_bounds__(256)
groupnorm_kernel(const float* __restrict__ x, const float* __restrict__ y,
                 const float* __restrict__ gn_scale, const float* __restrict__ gn_bias,
                 _Float16* __restrict__ hx, _Float16* __restrict__ hy) {
    const int b = blockIdx.x / NG;
    const int g = blockIdx.x % NG;
    const float* src = blockIdx.y ? y  : x;
    _Float16*    dst = blockIdx.y ? hy : hx;
    const float* base = src + ((size_t)b * CH + g * CG) * HW;

    const int nelem = CG * HW;  // 32768
    float s = 0.f, ss = 0.f;
    for (int i = threadIdx.x; i < nelem; i += 256) {
        float v = base[i];
        s += v;  ss += v * v;
    }
    __shared__ float red0[256], red1[256];
    red0[threadIdx.x] = s;  red1[threadIdx.x] = ss;
    __syncthreads();
#pragma unroll
    for (int off = 128; off > 0; off >>= 1) {
        if (threadIdx.x < off) {
            red0[threadIdx.x] += red0[threadIdx.x + off];
            red1[threadIdx.x] += red1[threadIdx.x + off];
        }
        __syncthreads();
    }
    const float inv_n = 1.0f / (float)nelem;
    const float mean  = red0[0] * inv_n;
    const float var   = red1[0] * inv_n - mean * mean;
    const float rstd  = rsqrtf(var + GN_EPS);

    for (int i = threadIdx.x; i < nelem; i += 256) {
        const int c = g * CG + (i >> 12);
        const int p = i & (HW - 1);
        float v = (base[i] - mean) * rstd * gn_scale[c] + gn_bias[c];
        dst[((size_t)b * HW + p) * CH + c] = (_Float16)v;
    }
}

// ---------------------------------------------------------------------------
// WMMA GEMM: Out[n, o] = sum_c A[n, c] * W[o, c] + bias[o]
// One wave: 16 rows x 256 cols (8 K-chunks x 16 col tiles = 128 WMMAs).
// TRANS=true stores [b][chan][pixel] (V projection, final projection).
// ---------------------------------------------------------------------------
template <bool TRANS, typename OT>
__global__ void __launch_bounds__(128)
gemm16_kernel(const _Float16* __restrict__ A, const _Float16* __restrict__ W16,
              const float* __restrict__ bias, OT* __restrict__ Out) {
    const int lane = threadIdx.x & 31;
    const int wid  = threadIdx.x >> 5;
    const int rt   = blockIdx.x * 4 + wid;   // row tile (16 rows)
    const int m    = lane & 15;
    const int hi   = lane >> 4;

    v8f acc[16];
#pragma unroll
    for (int t = 0; t < 16; ++t) acc[t] = vzero8f();

#pragma unroll
    for (int kc = 0; kc < 8; ++kc) {
        const _Float16* ap = A + (size_t)(rt * 16 + m) * CH + kc * 32 + hi * 16;
        v8h alo = *(const v8h*)ap;
        v8h ahi = *(const v8h*)(ap + 8);
        v16h a;
#pragma unroll
        for (int i = 0; i < 8; ++i) { a[i] = alo[i]; a[8 + i] = ahi[i]; }

#pragma unroll
        for (int ct = 0; ct < 16; ++ct) {
            v16h bmat = ld16h(W16 + (size_t)(ct * 16 + m) * CH + kc * 32 + hi * 16);
            acc[ct] = wmma_f16(a, bmat, acc[ct]);
        }
    }

#pragma unroll
    for (int ct = 0; ct < 16; ++ct) {
        const int   chan = ct * 16 + m;
        const float bv   = bias[chan];
#pragma unroll
        for (int v = 0; v < 8; ++v) {
            const int M    = v + hi * 8;
            const int grow = rt * 16 + M;
            const float val = acc[ct][v] + bv;
            if (TRANS) {
                const int bb = grow >> 12;
                const int p  = grow & (HW - 1);
                Out[((size_t)bb * CH + chan) * HW + p] = (OT)val;
            } else {
                Out[(size_t)grow * CH + chan] = (OT)val;
            }
        }
    }
}

// ---------------------------------------------------------------------------
// Flash attention with async double-buffered LDS staging of K/V.
//   Q : [B][HW][C] f16, K : [B][HW][C] f16, Vt: [B][C][HW] f16, O: [B][HW][C]
// Block = 4 waves = 4 consecutive 16-row query tiles of one batch; the block
// cooperatively DMA-stages each 32-key K/V block into LDS (async, ASYNCcnt)
// while computing on the previous block.
// ---------------------------------------------------------------------------
__global__ void __launch_bounds__(128)
attention_kernel(const _Float16* __restrict__ Q, const _Float16* __restrict__ K,
                 const _Float16* __restrict__ Vt, _Float16* __restrict__ O) {
    const int tid  = threadIdx.x;
    const int lane = tid & 31;
    const int wid  = tid >> 5;
    const int tile = blockIdx.x * 4 + wid;   // 0..2047
    const int b    = tile >> 8;              // 256 row tiles per batch
    const int rt   = tile & 255;
    const int m    = lane & 15;
    const int hi   = lane >> 4;

    const _Float16* Qb = Q  + (size_t)b * HW * CH;
    const _Float16* Kb = K  + (size_t)b * HW * CH;
    const _Float16* Vb = Vt + (size_t)b * CH * HW;
    _Float16*       Ob = O  + (size_t)b * HW * CH;

    __shared__ _Float16 Kbuf[2][32 * KSTRIDE];   // 2 x 16.5 KB
    __shared__ _Float16 Vbuf[2][CH * VSTRIDE];   // 2 x 20 KB
    __shared__ _Float16 Pbuf[4][16 * 32];        // 4 KB (C/D -> A layout bounce)
    _Float16* pbuf = Pbuf[wid];

    // ---- cooperative async stage of one 32-key block (K: 16KB, V: 16KB) ----
    const int kr = tid >> 2;                 // 0..31 (key row)
    const int kq = tid & 3;                  // 0..3  (128B quarter of the row)
    auto stage = [&](int j, int buf) {
        const _Float16* gk = Kb + (size_t)(j + kr) * CH + kq * 64;
        _Float16*       lk = &Kbuf[buf][kr * KSTRIDE + kq * 64];
#pragma unroll
        for (int i = 0; i < 8; ++i)
            async_g2l_b128(lk + i * 8, gk + i * 8);
#pragma unroll
        for (int rr = 0; rr < 2; ++rr) {
            const int c = tid * 2 + rr;
            const _Float16* gv = Vb + (size_t)c * HW + j;
            _Float16*       lv = &Vbuf[buf][c * VSTRIDE];
#pragma unroll
            for (int i = 0; i < 4; ++i)
                async_g2l_b128(lv + i * 8, gv + i * 8);
        }
    };

    // ---- load the 16x256 Q tile into registers (A-matrix layout) ----
    v16h qa[8];
#pragma unroll
    for (int kc = 0; kc < 8; ++kc) {
        const _Float16* qp = Qb + (size_t)(rt * 16 + m) * CH + kc * 32 + hi * 16;
        v8h lo = *(const v8h*)qp;
        v8h hh = *(const v8h*)(qp + 8);
        v16h a;
#pragma unroll
        for (int i = 0; i < 8; ++i) { a[i] = lo[i]; a[8 + i] = hh[i]; }
        qa[kc] = a;
    }

    v8f o_acc[16];
#pragma unroll
    for (int t = 0; t < 16; ++t) o_acc[t] = vzero8f();
    float m_state[8], l_state[8];
#pragma unroll
    for (int v = 0; v < 8; ++v) { m_state[v] = -3.0e38f; l_state[v] = 0.0f; }

    // prologue: stage block 0
    stage(0, 0);
    wait_async0();
    __syncthreads();

    const int NBLK = HW / 32;   // 128
    for (int jt = 0; jt < NBLK; ++jt) {
        const int cur = jt & 1;
        if (jt + 1 < NBLK) stage((jt + 1) * 32, cur ^ 1);   // overlap DMA w/ compute

        // ---- S = Q * K^T for 32 keys from LDS: two 16x16 f32 tiles ----
        v8f s0 = vzero8f(), s1 = vzero8f();
#pragma unroll
        for (int kc = 0; kc < 8; ++kc) {
            v16h b0 = ld16h(&Kbuf[cur][(m)      * KSTRIDE + kc * 32 + hi * 16]);
            v16h b1 = ld16h(&Kbuf[cur][(m + 16) * KSTRIDE + kc * 32 + hi * 16]);
            s0 = wmma_f16(qa[kc], b0, s0);
            s1 = wmma_f16(qa[kc], b1, s1);
        }

        // ---- online softmax over the 32 new scores per row ----
        float alpha[8];
#pragma unroll
        for (int v = 0; v < 8; ++v) {
            float a0 = s0[v] * ATT_SCALE;
            float a1 = s1[v] * ATT_SCALE;
            float r = fmaxf(a0, a1);
            r = fmaxf(r, __shfl_xor(r, 1, 32));
            r = fmaxf(r, __shfl_xor(r, 2, 32));
            r = fmaxf(r, __shfl_xor(r, 4, 32));
            r = fmaxf(r, __shfl_xor(r, 8, 32));
            const float mo = m_state[v];
            const float mn = fmaxf(mo, r);
            m_state[v] = mn;
            const float al = __expf(mo - mn);
            const float p0 = __expf(a0 - mn);
            const float p1 = __expf(a1 - mn);
            s0[v] = p0;  s1[v] = p1;
            float rs = p0 + p1;
            rs += __shfl_xor(rs, 1, 32);
            rs += __shfl_xor(rs, 2, 32);
            rs += __shfl_xor(rs, 4, 32);
            rs += __shfl_xor(rs, 8, 32);
            l_state[v] = l_state[v] * al + rs;
            alpha[v] = al;
        }

#pragma unroll
        for (int t = 0; t < 16; ++t)
#pragma unroll
            for (int v = 0; v < 8; ++v) o_acc[t][v] *= alpha[v];

        // ---- C/D layout -> A layout via per-wave LDS bounce (f16) ----
#pragma unroll
        for (int v = 0; v < 8; ++v) {
            const int M = v + hi * 8;
            pbuf[M * 32 + m]      = (_Float16)s0[v];
            pbuf[M * 32 + 16 + m] = (_Float16)s1[v];
        }
        v16h pa;
        {
            const _Float16* pr = pbuf + m * 32 + hi * 16;
#pragma unroll
            for (int i = 0; i < 8; ++i) { pa[i] = pr[i]; pa[8 + i] = pr[8 + i]; }
        }

        // ---- O += P * V from LDS (16 col tiles of 256 channels) ----
#pragma unroll
        for (int ct = 0; ct < 16; ++ct) {
            v16h bv = ld16h(&Vbuf[cur][(ct * 16 + m) * VSTRIDE + hi * 16]);
            o_acc[ct] = wmma_f16(pa, bv, o_acc[ct]);
        }

        wait_async0();      // our slice of next buffer landed
        __syncthreads();    // whole block done consuming cur + staging next
    }

    // ---- normalize and store O[b][row][c] ----
    float inv_l[8];
#pragma unroll
    for (int v = 0; v < 8; ++v) inv_l[v] = 1.0f / l_state[v];
#pragma unroll
    for (int ct = 0; ct < 16; ++ct)
#pragma unroll
        for (int v = 0; v < 8; ++v) {
            const int M = v + hi * 8;
            Ob[(size_t)(rt * 16 + M) * CH + ct * 16 + m] =
                (_Float16)(o_acc[ct][v] * inv_l[v]);
        }
}

// ---------------------------------------------------------------------------
// Host-side orchestration
// ---------------------------------------------------------------------------
extern "C" void kernel_launch(void* const* d_in, const int* in_sizes, int n_in,
                              void* d_out, int out_size, void* d_ws, size_t ws_size,
                              hipStream_t stream) {
    const float* x        = (const float*)d_in[0];
    const float* target   = (const float*)d_in[1];
    const float* gn_scale = (const float*)d_in[2];
    const float* gn_bias  = (const float*)d_in[3];
    const float* Wq = (const float*)d_in[4];   const float* bq = (const float*)d_in[5];
    const float* Wk = (const float*)d_in[6];   const float* bk = (const float*)d_in[7];
    const float* Wv = (const float*)d_in[8];   const float* bv = (const float*)d_in[9];
    const float* Wp = (const float*)d_in[10];  const float* bp = (const float*)d_in[11];
    float* out = (float*)d_out;

    const size_t NC = (size_t)NROWS * CH;
    _Float16* ws   = (_Float16*)d_ws;
    _Float16* hx   = ws;                       // later reused as attention output
    _Float16* hy   = ws + NC;
    _Float16* Qb   = ws + 2 * NC;
    _Float16* Kb   = ws + 3 * NC;
    _Float16* Vt   = ws + 4 * NC;
    _Float16* w16q = ws + 5 * NC;
    _Float16* w16k = w16q + 65536;
    _Float16* w16v = w16q + 2 * 65536;
    _Float16* w16p = w16q + 3 * 65536;
    _Float16* Oa   = hx;

    cvt_w_kernel<<<256, 256, 0, stream>>>(Wq, w16q, CH * CH);
    cvt_w_kernel<<<256, 256, 0, stream>>>(Wk, w16k, CH * CH);
    cvt_w_kernel<<<256, 256, 0, stream>>>(Wv, w16v, CH * CH);
    cvt_w_kernel<<<256, 256, 0, stream>>>(Wp, w16p, CH * CH);

    groupnorm_kernel<<<dim3(BATCH * NG, 2), 256, 0, stream>>>(
        x, target, gn_scale, gn_bias, hx, hy);

    gemm16_kernel<false, _Float16><<<512, 128, 0, stream>>>(hx, w16q, bq, Qb);
    gemm16_kernel<false, _Float16><<<512, 128, 0, stream>>>(hy, w16k, bk, Kb);
    gemm16_kernel<true,  _Float16><<<512, 128, 0, stream>>>(hy, w16v, bv, Vt);

    attention_kernel<<<512, 128, 0, stream>>>(Qb, Kb, Vt, Oa);

    gemm16_kernel<true, float><<<512, 128, 0, stream>>>(Oa, w16p, bp, out);
}
